// CrossAttention_85040352461007
// MI455X (gfx1250) — matrix operand order repaired and checked by reading, not compiled, not run on hardware
//
#include <hip/hip_runtime.h>
#include <hip/hip_fp16.h>

typedef __attribute__((ext_vector_type(16))) _Float16 v16h;
typedef __attribute__((ext_vector_type(8)))  float    v8f;

#define WMMA_F32_F16(a, b, c) \
  __builtin_amdgcn_wmma_f32_16x16x32_f16(false, (a), false, (b), (short)0, (c), false, false)

// ---------------------------------------------------------------------------
// gfx1250 async global->LDS copy (ASYNCcnt-tracked), per-lane 16B.
// VDST = LDS byte offset (low 32 bits of generic shared address);
// VADDR = 64-bit global address; SADDR = off.
// ---------------------------------------------------------------------------
__device__ __forceinline__ void async_cp16(const _Float16* g, _Float16* l) {
  const unsigned lds = (unsigned)(uintptr_t)l;
  const unsigned long long ga = (unsigned long long)(uintptr_t)g;
  asm volatile("global_load_async_to_lds_b128 %0, %1, off"
               :: "v"(lds), "v"(ga) : "memory");
}
__device__ __forceinline__ void async_wait0() {
  asm volatile("s_wait_asynccnt 0" ::: "memory");
}

// ---------------------------------------------------------------------------
// Fragment loaders (contiguous k order: element s of half h is k = 16*h + s), sources are row-major LDS tiles.
// ---------------------------------------------------------------------------

// A-matrix 16x32 f16: lane m = lane&15; VGPR v holds K = 2*v + 16*half
__device__ __forceinline__ v16h ldsA(const _Float16* base, int stride) {
  const int lane = threadIdx.x & 31;
  const int m = lane & 15, hf = lane >> 4;
  v16h a;
#pragma unroll
  for (int v = 0; v < 8; ++v) {
    const int k = (hf << 4) + (v << 1);
    a[2 * v]     = base[m * stride + k];
    a[2 * v + 1] = base[m * stride + k + 1];
  }
  return a;
}

// B-matrix 32x16, source stored [n][k] (e.g. rows of W give B[k][n] = W[n][k])
__device__ __forceinline__ v16h ldsB_nk(const _Float16* base, int stride) {
  const int lane = threadIdx.x & 31;
  const int n = lane & 15, hf = lane >> 4;
  v16h b;
#pragma unroll
  for (int j = 0; j < 8; ++j) {
    const int k = (hf << 4) + (j << 1);
    b[2 * j]     = base[n * stride + k];
    b[2 * j + 1] = base[n * stride + k + 1];
  }
  return b;
}

// B-matrix 32x16, source stored [k][n] (e.g. V tile: B[k][n] = V[k][n])
__device__ __forceinline__ v16h ldsB_kn(const _Float16* base, int stride) {
  const int lane = threadIdx.x & 31;
  const int n = lane & 15, hf = lane >> 4;
  v16h b;
#pragma unroll
  for (int j = 0; j < 8; ++j) {
    const int k = (hf << 4) + (j << 1);
    b[2 * j]     = base[k * stride + n];
    b[2 * j + 1] = base[(k + 1) * stride + n];
  }
  return b;
}

// ---------------------------------------------------------------------------
// Generic WMMA GEMM: Out = X @ W^T + bias.  X:[M,K] (f32 or f16), W:[N,K] f32.
// Block tile 128x128, 8 waves (4m x 2n), each wave 32x64 (2x4 wmma frags).
// Software pipelined: VGPR prefetch of K-step k+1 overlaps WMMAs on k;
// double-buffered LDS, one barrier per K-step.
// HEADSPLIT: f16 store scattered to [B, N/64, S, 64]; else f32 row-major.
// ---------------------------------------------------------------------------
template <int XF16, int HEADSPLIT>
__global__ __launch_bounds__(256) void wmma_gemm_xwT(
    const void* __restrict__ Xv, const float* __restrict__ W,
    const float* __restrict__ bias, void* __restrict__ Outv,
    int M, int N, int K, int S) {
  __shared__ _Float16 As[2][128][40];
  __shared__ _Float16 Bs[2][128][40];
  const int tid = threadIdx.x;
  const int lane = tid & 31, wave = tid >> 5;
  const int waveM = wave & 3, waveN = wave >> 2;
  const int gm0 = blockIdx.y * 128, gn0 = blockIdx.x * 128;
  const int kc = (tid & 7) * 4, rbase = tid >> 3;
  const v8f zf = {0.f, 0.f, 0.f, 0.f, 0.f, 0.f, 0.f, 0.f};
  v8f acc[2][4];
#pragma unroll
  for (int mt = 0; mt < 2; ++mt)
#pragma unroll
    for (int nt = 0; nt < 4; ++nt) acc[mt][nt] = zf;

  const float*    Xf = (const float*)Xv;
  const _Float16* Xh = (const _Float16*)Xv;

  float4    ra[4], rb[4];   // f32 prefetch regs
  _Float16  rah[4][4];      // f16 prefetch regs (XF16 path)

  auto stage_load = [&](int k0) {
#pragma unroll
    for (int p = 0; p < 4; ++p) {
      const int row = p * 32 + rbase;
      if (XF16) {
        const _Float16* xp = &Xh[(size_t)(gm0 + row) * K + k0 + kc];
#pragma unroll
        for (int j = 0; j < 4; ++j) rah[p][j] = xp[j];
      } else {
        ra[p] = *(const float4*)&Xf[(size_t)(gm0 + row) * K + k0 + kc];
      }
      rb[p] = *(const float4*)&W[(size_t)(gn0 + row) * K + k0 + kc];
    }
  };
  auto stage_store = [&](int b) {
#pragma unroll
    for (int p = 0; p < 4; ++p) {
      const int row = p * 32 + rbase;
      if (XF16) {
#pragma unroll
        for (int j = 0; j < 4; ++j) As[b][row][kc + j] = rah[p][j];
      } else {
        As[b][row][kc + 0] = (_Float16)ra[p].x; As[b][row][kc + 1] = (_Float16)ra[p].y;
        As[b][row][kc + 2] = (_Float16)ra[p].z; As[b][row][kc + 3] = (_Float16)ra[p].w;
      }
      Bs[b][row][kc + 0] = (_Float16)rb[p].x; Bs[b][row][kc + 1] = (_Float16)rb[p].y;
      Bs[b][row][kc + 2] = (_Float16)rb[p].z; Bs[b][row][kc + 3] = (_Float16)rb[p].w;
    }
  };

  const int nk = K / 32;
  stage_load(0);
  stage_store(0);
  __syncthreads();
  int buf = 0;
  for (int ks = 0; ks < nk; ++ks) {
    if (ks + 1 < nk) stage_load((ks + 1) * 32);  // loads in flight during WMMA

    v16h af[2], bf[4];
#pragma unroll
    for (int mt = 0; mt < 2; ++mt)
      af[mt] = ldsA(&As[buf][waveM * 32 + mt * 16][0], 40);
#pragma unroll
    for (int nt = 0; nt < 4; ++nt)
      bf[nt] = ldsB_nk(&Bs[buf][waveN * 64 + nt * 16][0], 40);
#pragma unroll
    for (int mt = 0; mt < 2; ++mt)
#pragma unroll
      for (int nt = 0; nt < 4; ++nt)
        acc[mt][nt] = WMMA_F32_F16(af[mt], bf[nt], acc[mt][nt]);

    if (ks + 1 < nk) stage_store(buf ^ 1);  // s_wait_loadcnt lands here
    __syncthreads();
    buf ^= 1;
  }

  // Epilogue: bias add + store. C layout: row m = r + 8*half, col n = lane&15.
  const int hf = lane >> 4, n0 = lane & 15;
  const int nH = N >> 6;
#pragma unroll
  for (int mt = 0; mt < 2; ++mt)
#pragma unroll
    for (int nt = 0; nt < 4; ++nt)
#pragma unroll
      for (int r = 0; r < 8; ++r) {
        const int gm = gm0 + waveM * 32 + mt * 16 + r + 8 * hf;
        const int gn = gn0 + waveN * 64 + nt * 16 + n0;
        const float val = acc[mt][nt][r] + bias[gn];
        if (HEADSPLIT) {
          const int bb = gm / S, sr = gm - bb * S;
          const int hh = gn >> 6, dk = gn & 63;
          ((_Float16*)Outv)[((((size_t)bb * nH + hh) * S + sr) << 6) + dk] =
              (_Float16)val;
        } else {
          ((float*)Outv)[(size_t)gm * N + gn] = val;
        }
      }
}

// ---------------------------------------------------------------------------
// Flash attention: grid (NQ/64, B*H), block 128 (4 waves).
// Each wave owns 16 query rows x DK=64. K/V tiles stream through
// double-buffered LDS via global_load_async_to_lds_b128, prefetching the
// next tile while WMMAs run on the current one.
// ---------------------------------------------------------------------------
__global__ __launch_bounds__(128) void flash_attn(
    const _Float16* __restrict__ Qh, const _Float16* __restrict__ Kh,
    const _Float16* __restrict__ Vh, _Float16* __restrict__ Yatt,
    int Hn, int NQn, int Pn) {
  __shared__ _Float16 Qs[64][72];
  __shared__ _Float16 Ks[2][64][72];
  __shared__ _Float16 Vs[2][64][72];
  __shared__ _Float16 Ps[4][16][72];  // per-wave C-layout -> A-layout restage

  const int tid = threadIdx.x, lane = tid & 31, wave = tid >> 5;
  const int bh = blockIdx.y, b = bh / Hn, h = bh - b * Hn;
  const int q0 = blockIdx.x * 64;
  const _Float16* Qbase = Qh + (((size_t)b * Hn + h) * NQn + q0) * 64;
  const _Float16* Kbase = Kh + ((size_t)b * Hn + h) * Pn * 64;
  const _Float16* Vbase = Vh + ((size_t)b * Hn + h) * Pn * 64;

  const int row = tid >> 1, c0 = (tid & 1) * 32;
  auto issue_kv = [&](int kt, int bufi) {
#pragma unroll
    for (int j = 0; j < 4; ++j) {
      async_cp16(Kbase + (size_t)(kt + row) * 64 + c0 + j * 8,
                 &Ks[bufi][row][c0 + j * 8]);
      async_cp16(Vbase + (size_t)(kt + row) * 64 + c0 + j * 8,
                 &Vs[bufi][row][c0 + j * 8]);
    }
  };

  // Q tile + first K/V tile, async.
#pragma unroll
  for (int j = 0; j < 4; ++j)
    async_cp16(Qbase + row * 64 + c0 + j * 8, &Qs[row][c0 + j * 8]);
  issue_kv(0, 0);
  async_wait0();
  __syncthreads();

  const v16h aq0 = ldsA(&Qs[wave * 16][0], 72);
  const v16h aq1 = ldsA(&Qs[wave * 16][32], 72);

  const v8f zf = {0.f, 0.f, 0.f, 0.f, 0.f, 0.f, 0.f, 0.f};
  v8f acc[4];
#pragma unroll
  for (int nt = 0; nt < 4; ++nt) acc[nt] = zf;
  float mrow[8], lrow[8];
#pragma unroll
  for (int r = 0; r < 8; ++r) { mrow[r] = -1.0e30f; lrow[r] = 0.f; }
  const int hf = lane >> 4, n0 = lane & 15;

  int buf = 0;
  for (int kt = 0; kt < Pn; kt += 64) {
    if (kt) {            // tile(buf) staged by asyncs issued last iteration
      async_wait0();
      __syncthreads();   // also: everyone done reading buf^1 -> safe to refill
    }
    if (kt + 64 < Pn) issue_kv(kt + 64, buf ^ 1);  // overlap with compute

    // S = (Q @ K^T) * 1/sqrt(64); B[k][n] = K[n][k] -> nk loader
    v8f s[4];
#pragma unroll
    for (int nt = 0; nt < 4; ++nt) {
      const v16h bk0 = ldsB_nk(&Ks[buf][nt * 16][0], 72);
      const v16h bk1 = ldsB_nk(&Ks[buf][nt * 16][32], 72);
      v8f t = WMMA_F32_F16(aq0, bk0, zf);
      s[nt] = WMMA_F32_F16(aq1, bk1, t);
#pragma unroll
      for (int r = 0; r < 8; ++r) s[nt][r] *= 0.125f;
    }

    // Online softmax: row m = r + 8*half spans the 16 lanes of this half-wave.
    float tmax[8];
#pragma unroll
    for (int r = 0; r < 8; ++r)
      tmax[r] = fmaxf(fmaxf(s[0][r], s[1][r]), fmaxf(s[2][r], s[3][r]));
#pragma unroll
    for (int off = 8; off; off >>= 1)
#pragma unroll
      for (int r = 0; r < 8; ++r)
        tmax[r] = fmaxf(tmax[r], __shfl_xor(tmax[r], off, 32));
    float corr[8];
#pragma unroll
    for (int r = 0; r < 8; ++r) {
      const float nm = fmaxf(mrow[r], tmax[r]);
      corr[r] = __expf(mrow[r] - nm);
      mrow[r] = nm;
    }
    float rs[8];
#pragma unroll
    for (int r = 0; r < 8; ++r) rs[r] = 0.f;
#pragma unroll
    for (int nt = 0; nt < 4; ++nt)
#pragma unroll
      for (int r = 0; r < 8; ++r) {
        const float p = __expf(s[nt][r] - mrow[r]);
        s[nt][r] = p;
        rs[r] += p;
      }
#pragma unroll
    for (int off = 8; off; off >>= 1)
#pragma unroll
      for (int r = 0; r < 8; ++r) rs[r] += __shfl_xor(rs[r], off, 32);
#pragma unroll
    for (int r = 0; r < 8; ++r) lrow[r] = lrow[r] * corr[r] + rs[r];
#pragma unroll
    for (int nt = 0; nt < 4; ++nt)
#pragma unroll
      for (int r = 0; r < 8; ++r) acc[nt][r] *= corr[r];

    // Restage P (C layout) into LDS, reload as A fragments (wave-local region).
#pragma unroll
    for (int nt = 0; nt < 4; ++nt)
#pragma unroll
      for (int r = 0; r < 8; ++r)
        Ps[wave][r + 8 * hf][nt * 16 + n0] = (_Float16)s[nt][r];
    const v16h ap0 = ldsA(&Ps[wave][0][0], 72);
    const v16h ap1 = ldsA(&Ps[wave][0][32], 72);

    // acc += P @ V;  B[k][n] = V[k][dk0+n] -> kn loader
#pragma unroll
    for (int nt = 0; nt < 4; ++nt) {
      const v16h bv0 = ldsB_kn(&Vs[buf][0][nt * 16], 72);
      const v16h bv1 = ldsB_kn(&Vs[buf][32][nt * 16], 72);
      v8f t = WMMA_F32_F16(ap0, bv0, acc[nt]);
      acc[nt] = WMMA_F32_F16(ap1, bv1, t);
    }
    buf ^= 1;
  }

  // Normalize and merge heads into Yatt [B*NQ, H*64] f16.
  const int Dm = Hn * 64;
#pragma unroll
  for (int nt = 0; nt < 4; ++nt)
#pragma unroll
    for (int r = 0; r < 8; ++r) {
      const int gq = q0 + wave * 16 + r + 8 * hf;
      const int col = h * 64 + nt * 16 + n0;
      Yatt[((size_t)b * NQn + gq) * Dm + col] = (_Float16)(acc[nt][r] / lrow[r]);
    }
}

// ---------------------------------------------------------------------------
// Row LayerNorm over D=1024, in-place safe (block owns its row).
// ---------------------------------------------------------------------------
__global__ __launch_bounds__(256) void ln_rows(const float* __restrict__ Y,
                                               const float* __restrict__ w,
                                               const float* __restrict__ bb,
                                               float* __restrict__ out, int D) {
  __shared__ float red[256];
  const int row = blockIdx.x, tid = threadIdx.x;
  float x[4];
  float s = 0.f;
#pragma unroll
  for (int j = 0; j < 4; ++j) {
    x[j] = Y[(size_t)row * D + tid + j * 256];
    s += x[j];
  }
  red[tid] = s;
  __syncthreads();
  for (int off = 128; off; off >>= 1) {
    if (tid < off) red[tid] += red[tid + off];
    __syncthreads();
  }
  const float mu = red[0] / (float)D;
  __syncthreads();
  float v = 0.f;
#pragma unroll
  for (int j = 0; j < 4; ++j) {
    const float d = x[j] - mu;
    v += d * d;
  }
  red[tid] = v;
  __syncthreads();
  for (int off = 128; off; off >>= 1) {
    if (tid < off) red[tid] += red[tid + off];
    __syncthreads();
  }
  const float rstd = rsqrtf(red[0] / (float)D + 1e-5f);
#pragma unroll
  for (int j = 0; j < 4; ++j) {
    const int c = tid + j * 256;
    out[(size_t)row * D + c] = (x[j] - mu) * rstd * w[c] + bb[c];
  }
}

// ---------------------------------------------------------------------------
extern "C" void kernel_launch(void* const* d_in, const int* in_sizes, int n_in,
                              void* d_out, int out_size, void* d_ws,
                              size_t ws_size, hipStream_t stream) {
  (void)in_sizes; (void)n_in; (void)out_size; (void)ws_size;
  const float* q   = (const float*)d_in[0];
  const float* C   = (const float*)d_in[1];
  const float* Wq  = (const float*)d_in[2];
  const float* bq  = (const float*)d_in[3];
  const float* Wk  = (const float*)d_in[4];
  const float* bk  = (const float*)d_in[5];
  const float* Wv  = (const float*)d_in[6];
  const float* bv  = (const float*)d_in[7];
  const float* Wo  = (const float*)d_in[8];
  const float* bo  = (const float*)d_in[9];
  const float* lnw = (const float*)d_in[10];
  const float* lnb = (const float*)d_in[11];
  float* out = (float*)d_out;

  const int Bn = 4, NQ = 1024, P = 2048, D = 1024, H = 16;

  char* ws = (char*)d_ws;
  _Float16* Qh = (_Float16*)(ws);                        // 8  MB [B,H,NQ,64]
  _Float16* Kh = (_Float16*)(ws + ((size_t)8 << 20));    // 16 MB [B,H,P,64]
  _Float16* Vh = (_Float16*)(ws + ((size_t)24 << 20));   // 16 MB [B,H,P,64]
  _Float16* Ya = (_Float16*)(ws + ((size_t)40 << 20));   // 8  MB [B*NQ,D]

  const dim3 blk(256);
  // Q/K/V projections with fused head-split f16 store
  wmma_gemm_xwT<0, 1><<<dim3(D / 128, (Bn * NQ) / 128), blk, 0, stream>>>(
      q, Wq, bq, Qh, Bn * NQ, D, D, NQ);
  wmma_gemm_xwT<0, 1><<<dim3(D / 128, (Bn * P) / 128), blk, 0, stream>>>(
      C, Wk, bk, Kh, Bn * P, D, D, P);
  wmma_gemm_xwT<0, 1><<<dim3(D / 128, (Bn * P) / 128), blk, 0, stream>>>(
      C, Wv, bv, Vh, Bn * P, D, D, P);
  // Flash attention
  flash_attn<<<dim3(NQ / 64, Bn * H), dim3(128), 0, stream>>>(
      Qh, Kh, Vh, Ya, H, NQ, P);
  // Output projection (f16 input) -> f32 straight into d_out
  wmma_gemm_xwT<1, 0><<<dim3(D / 128, (Bn * NQ) / 128), blk, 0, stream>>>(
      Ya, Wo, bo, out, Bn * NQ, D, D, NQ);
  // LayerNorm in place
  ln_rows<<<dim3(Bn * NQ), dim3(256), 0, stream>>>(out, lnw, lnb, out, D);
}
